// LowRankProcessNeurons_15668040696247
// MI455X (gfx1250) — matrix operand order repaired and checked by reading, not compiled, not run on hardware
//
#include <hip/hip_runtime.h>
#include <cmath>

#define USE_ASYNC_STAGING 1   // set 0 to revert to VGPR-staged global->LDS copies

// ---------------------------------------------------------------------------
// Problem constants (B=4, S=1024, H=1024, N_IN=64, N_PROC=16, R=128)
// ---------------------------------------------------------------------------
#define BSZ   4
#define SEQ   1024
#define HD    1024
#define NIN   64
#define NP    16
#define RR    128
#define BS    (BSZ * SEQ)          // 4096 rows
#define NCOL  (NP * RR)            // 2048 inner columns

typedef __attribute__((ext_vector_type(16))) _Float16 v16h;
typedef __attribute__((ext_vector_type(4)))  _Float16 v4h;
typedef __attribute__((ext_vector_type(8)))  float    v8f;

union FragH { v16h h; uint4 q[2]; };

#define KT    32      // K-tile
#define TSTR  40      // LDS row stride in halves: 80B rows, 16B aligned, conflict-free frags
#define ABUF  4096    // A double-buffer stride (halves, pow2 -> XOR flip), 2560 used
#define BBUF  16384   // B double-buffer stride (halves, pow2 -> XOR flip), 10240 used

// ---------------------------------------------------------------------------
// CDNA5 async global->LDS DMA (ASYNCcnt-tracked), per ISA §15.18.3 op 98.
// VDST VGPR carries the LDS byte offset (generic LDS pointer low 32 bits).
// ---------------------------------------------------------------------------
__device__ __forceinline__ void async_load_b128(unsigned lds_off, const void* g) {
  asm volatile("global_load_async_to_lds_b128 %0, %1, off"
               :: "v"(lds_off), "v"((unsigned long long)(uintptr_t)g)
               : "memory");
}
__device__ __forceinline__ void wait_async0() {
#if __has_builtin(__builtin_amdgcn_s_wait_asynccnt)
  __builtin_amdgcn_s_wait_asynccnt(0);
#else
  asm volatile("s_wait_asynccnt 0x0" ::: "memory");
#endif
}
__device__ __forceinline__ unsigned lds_off32(const void* p) {
  return (unsigned)(uintptr_t)p;   // LDS aperture: addr[31:0] = LDS byte offset
}

// ---------------------------------------------------------------------------
// Elementwise f32 -> f16 (4 elements / thread) -- used for X
// ---------------------------------------------------------------------------
__global__ __launch_bounds__(256)
void cvt_f16_kernel(const float* __restrict__ in, _Float16* __restrict__ out, int n4) {
  int i = blockIdx.x * 256 + threadIdx.x;
  if (i >= n4) return;
  float4 v = ((const float4*)in)[i];
  v4h h = { (_Float16)v.x, (_Float16)v.y, (_Float16)v.z, (_Float16)v.w };
  *(v4h*)&out[4 * i] = h;
}

// ---------------------------------------------------------------------------
// Batched tiled transpose + f32->f16 convert (coalesced in and out).
//   Wd: down_proj (NP,H,R) -> Wdt[col=n*128+r][k=h]   (col-major B for GEMM1)
//   Wu: up_proj   (NP,R,H) -> Wut[col=h][k=n*128+r]   (col-major B for GEMM2)
// ---------------------------------------------------------------------------
__global__ __launch_bounds__(256)
void transpose_cvt_kernel(const float* __restrict__ in, _Float16* __restrict__ out,
                          int rowsIn, int colsIn, int outStride, int outBatchStride) {
  __shared__ float tile[32][33];
  const int n  = blockIdx.z;
  const float* inb = in + (size_t)n * rowsIn * colsIn;
  _Float16* outb = out + (size_t)n * outBatchStride;
  const int c0 = blockIdx.x * 32;
  const int r0 = blockIdx.y * 32;
  const int tx = threadIdx.x & 31;
  const int ty = threadIdx.x >> 5;        // 0..7
  #pragma unroll
  for (int i = 0; i < 32; i += 8)
    tile[ty + i][tx] = inb[(size_t)(r0 + ty + i) * colsIn + (c0 + tx)];
  __syncthreads();
  #pragma unroll
  for (int i = 0; i < 32; i += 8)
    outb[(size_t)(c0 + ty + i) * outStride + (r0 + tx)] = (_Float16)tile[tx][ty + i];
}

// ---------------------------------------------------------------------------
// Gates: conv (kernel 5 over S, reduce N_IN) + bias + sigmoid
// ---------------------------------------------------------------------------
__global__ __launch_bounds__(256)
void gates_kernel(const float* __restrict__ enr, const float* __restrict__ cw,
                  const float* __restrict__ cb, float* __restrict__ gout) {
  int gid = blockIdx.x * 256 + threadIdx.x;   // 65536
  int n  = gid & 15;
  int bs = gid >> 4;
  int b  = bs >> 10;
  int s  = bs & 1023;
  float acc = cb[n];
  #pragma unroll
  for (int dk = 0; dk < 5; ++dk) {
    int ss = s + dk - 2;
    if (ss < 0 || ss >= SEQ) continue;
    const float4* e4 = (const float4*)&enr[((size_t)b * SEQ + ss) * NIN];
    const float4* w4 = (const float4*)&cw[n * 5 * NIN + dk * NIN];
    #pragma unroll
    for (int ci = 0; ci < NIN / 4; ++ci) {
      float4 e = e4[ci], w = w4[ci];
      acc += e.x * w.x + e.y * w.y + e.z * w.z + e.w * w.w;
    }
  }
  gout[gid] = 1.0f / (1.0f + expf(-acc));
}

// ---------------------------------------------------------------------------
// Tiled WMMA GEMM:  C(M x N) = A(M x K) @ B(K x N),  f16 in, f32 accumulate.
//   A  : row-major (k contiguous), stride KK;  Bt : column-major, stride KK
// Workgroup: 256 threads = 8 waves, tile 64(M) x 256(N); wave tile 32x64 =
// 8 accumulators. Double-buffered LDS (XOR-flip pow2 offsets), k-loop rolled
// (#pragma unroll 1) so accs stay in fixed regs, final K-step peeled.
// Staging: async global->LDS DMA (ASYNCcnt) -- no VGPR round trip.
// GATED: per-column expert gate folded in, f16 store to outH; else f32 to outF.
// ---------------------------------------------------------------------------
template<int KK, int NN, bool GATED>
__global__ __launch_bounds__(256, 1)
void gemm_wmma_kernel(const _Float16* __restrict__ A, const _Float16* __restrict__ Bt,
                      const float* __restrict__ gates,
                      _Float16* __restrict__ outH, float* __restrict__ outF) {
  __shared__ _Float16 AsF[2 * ABUF];       // per buf: 64 rows x 32 k (TSTR=40)
  __shared__ _Float16 BsF[2 * BBUF];       // per buf: 256 cols x 32 k (col-major tile)
  __shared__ float    gLds[64 * NP];

  const int tid = threadIdx.x;
  const int l   = tid & 31;
  const int w   = tid >> 5;
  const int wm  = w & 1;                   // 2 m sub-tiles of 32
  const int wn  = w >> 1;                  // 4 n sub-tiles of 64
  const int m0  = blockIdx.y * 64;
  const int n0  = blockIdx.x * 256;
  const int lm  = l & 15;
  const int lhi = l >> 4;

  v8f acc[2][4] = {};

  // Staging map: thread t handles (row = t>>2, kchunk = (t&3)*8 halves).
  const int sRow = tid >> 2;               // 0..63
  const int sKc  = (tid & 3) * 8;          // 0..24
  const _Float16* aG = &A[(size_t)(m0 + sRow) * KK + sKc];
  const _Float16* bG = &Bt[(size_t)(n0 + sRow) * KK + sKc];   // + it*64 cols

  // Per-lane LDS fragment indices (halves), XOR-flipped each K-step.
  int aIdx  = (wm * 32 + lm) * TSTR + lhi * 8;        // A frag base (mi: +16*TSTR)
  int bIdx  = (wn * 64 + lm) * TSTR + lhi * 8;        // B frag base (j:  +16*TSTR)
  // Staging LDS byte offsets (first in-loop stage -> buffer 1).
  unsigned sAOff = lds_off32(&AsF[sRow * TSTR + sKc + ABUF]);
  unsigned sBOff = lds_off32(&BsF[sRow * TSTR + sKc + BBUF]);

  // Gate preload for the epilogue -- tiny, once.
  if (GATED) {
    #pragma unroll
    for (int i = 0; i < 4; ++i) {
      int idx = tid + i * 256;             // 0..1023 = 64 rows x 16 experts
      gLds[idx] = gates[(size_t)(m0 + (idx >> 4)) * NP + (idx & 15)];
    }
  }

  // One K-step: load all 6 fragments, then a burst of 8 WMMAs.
  auto compute = [&](int aOff, int bOff) {
    FragH a[2], b[4];
    #pragma unroll
    for (int mi = 0; mi < 2; ++mi) {
      const _Float16* ap = AsF + aOff + mi * 16 * TSTR;
      a[mi].q[0] = *(const uint4*)(ap);
      a[mi].q[1] = *(const uint4*)(ap + 16);
    }
    #pragma unroll
    for (int j = 0; j < 4; ++j) {
      const _Float16* bp = BsF + bOff + j * 16 * TSTR;
      b[j].q[0] = *(const uint4*)(bp);
      b[j].q[1] = *(const uint4*)(bp + 16);
    }
    #pragma unroll
    for (int j = 0; j < 4; ++j) {
      acc[0][j] = __builtin_amdgcn_wmma_f32_16x16x32_f16(
          false, a[0].h, false, b[j].h, (short)0, acc[0][j], false, false);
      acc[1][j] = __builtin_amdgcn_wmma_f32_16x16x32_f16(
          false, a[1].h, false, b[j].h, (short)0, acc[1][j], false, false);
    }
  };

#if USE_ASYNC_STAGING
  // Stage one K-tile via async DMA into the buffer addressed by sAOff/sBOff^flip.
  auto stage_async = [&](int k0, unsigned aOffB, unsigned bOffB) {
    async_load_b128(aOffB, aG + k0);
    #pragma unroll
    for (int it = 0; it < 4; ++it)
      async_load_b128(bOffB + it * (64 * TSTR * 2), bG + (size_t)it * 64 * KK + k0);
  };

  // Prologue: async-stage tile 0 into buffer 0.
  stage_async(0, sAOff ^ (ABUF * 2), sBOff ^ (BBUF * 2));

  constexpr int NT = KK / KT;
  #pragma unroll 1
  for (int kt = 0; kt < NT - 1; ++kt) {
    wait_async0();                         // own async DMAs done (before barrier!)
    __syncthreads();                       // => everyone's DMAs to cur buf done

    stage_async(kt * KT + KT, sAOff, sBOff);   // DMA next tile -> other buffer
    __builtin_prefetch(aG + kt * KT + 2 * KT, 0, 3);
    __builtin_prefetch(bG + kt * KT + 2 * KT, 0, 3);

    compute(aIdx, bIdx);

    aIdx  ^= ABUF;       bIdx  ^= BBUF;    // flip compute buffer
    sAOff ^= ABUF * 2;   sBOff ^= BBUF * 2;  // flip stage buffer (byte offsets)
  }
  wait_async0();
  __syncthreads();
  compute(aIdx, bIdx);                     // peeled final K-step
#else
  int sAIdx = sRow * TSTR + sKc + ABUF;
  int sBIdx = sRow * TSTR + sKc + BBUF;
  {
    uint4 a0 = *(const uint4*)(aG);
    uint4 b0[4];
    #pragma unroll
    for (int it = 0; it < 4; ++it)
      b0[it] = *(const uint4*)(bG + (size_t)it * 64 * KK);
    *(uint4*)&AsF[sRow * TSTR + sKc] = a0;
    #pragma unroll
    for (int it = 0; it < 4; ++it)
      *(uint4*)&BsF[(sRow + it * 64) * TSTR + sKc] = b0[it];
  }
  constexpr int NT = KK / KT;
  #pragma unroll 1
  for (int kt = 0; kt < NT - 1; ++kt) {
    __syncthreads();
    const int k0 = kt * KT;
    uint4 an = *(const uint4*)(aG + k0 + KT);
    uint4 bn[4];
    #pragma unroll
    for (int it = 0; it < 4; ++it)
      bn[it] = *(const uint4*)(bG + (size_t)it * 64 * KK + k0 + KT);
    __builtin_prefetch(aG + k0 + 2 * KT, 0, 3);
    __builtin_prefetch(bG + k0 + 2 * KT, 0, 3);
    compute(aIdx, bIdx);
    *(uint4*)&AsF[sAIdx] = an;
    #pragma unroll
    for (int it = 0; it < 4; ++it)
      *(uint4*)&BsF[sBIdx + it * 64 * TSTR] = bn[it];
    aIdx  ^= ABUF;  bIdx  ^= BBUF;
    sAIdx ^= ABUF;  sBIdx ^= BBUF;
  }
  __syncthreads();
  compute(aIdx, bIdx);
#endif

  // ---- epilogue: C/D layout lane(l&15)=N, VGPR i -> M=(l>>4)*8+i ----
  #pragma unroll
  for (int mi = 0; mi < 2; ++mi) {
    const int rbase = m0 + wm * 32 + mi * 16 + lhi * 8;
    #pragma unroll
    for (int j = 0; j < 4; ++j) {
      const int col = n0 + wn * 64 + j * 16 + lm;
      if (GATED) {
        const int e = (n0 + wn * 64 + j * 16) >> 7;  // wave-uniform expert id
        #pragma unroll
        for (int i = 0; i < 8; ++i) {
          const int row = rbase + i;
          float g = gLds[(row - m0) * NP + e];
          outH[(size_t)row * NN + col] = (_Float16)(acc[mi][j][i] * g);
        }
      } else {
        #pragma unroll
        for (int i = 0; i < 8; ++i)
          outF[(size_t)(rbase + i) * NN + col] = acc[mi][j][i];
      }
    }
  }
}

// ---------------------------------------------------------------------------
// Launch
// ---------------------------------------------------------------------------
extern "C" void kernel_launch(void* const* d_in, const int* in_sizes, int n_in,
                              void* d_out, int out_size, void* d_ws, size_t ws_size,
                              hipStream_t stream) {
  const float* intermediate = (const float*)d_in[0];   // (B,S,H)
  const float* enriched     = (const float*)d_in[1];   // (B,S,N_IN)
  const float* conv_w       = (const float*)d_in[2];   // (NP,1,5,N_IN)
  const float* conv_b       = (const float*)d_in[3];   // (NP)
  const float* down_proj    = (const float*)d_in[4];   // (NP,H,R)
  const float* up_proj      = (const float*)d_in[5];   // (NP,R,H)

  float* out   = (float*)d_out;                        // output (B,S,H) f32
  float* gates = out + (size_t)BS * HD;                // process_activations (B,S,NP)

  char* ws = (char*)d_ws;
  _Float16* Xh  = (_Float16*)(ws);                     //  8 MB: X f16 (4096x1024) row-major
  _Float16* Wdt = (_Float16*)(ws + (size_t)(8  << 20));//  4 MB: Wd f16 col-major (2048 x 1024k)
  _Float16* Wut = (_Float16*)(ws + (size_t)(12 << 20));//  4 MB: Wu f16 col-major (1024 x 2048k)
  _Float16* Dg  = (_Float16*)(ws + (size_t)(16 << 20));// 16 MB: gated down f16 (4096x2048)

  // 1) f16 converts / weight transposes (one-time layout passes)
  cvt_f16_kernel<<<(BS * HD / 4) / 256, 256, 0, stream>>>(intermediate, Xh, BS * HD / 4);
  {
    dim3 grid(RR / 32, HD / 32, NP);                   // down_proj: 1024x128 per expert
    transpose_cvt_kernel<<<grid, 256, 0, stream>>>(down_proj, Wdt, HD, RR, HD, RR * HD);
  }
  {
    dim3 grid(HD / 32, RR / 32, NP);                   // up_proj: 128x1024 per expert
    transpose_cvt_kernel<<<grid, 256, 0, stream>>>(up_proj, Wut, RR, HD, NCOL, RR);
  }

  // 2) gates (conv5 + sigmoid), also the second reference output
  gates_kernel<<<(BS * NP) / 256, 256, 0, stream>>>(enriched, conv_w, conv_b, gates);

  // 3) GEMM1: Dg = gate * (X @ Wd)   [f16 out, K=1024, N=2048]
  {
    dim3 grid(NCOL / 256, BS / 64);
    gemm_wmma_kernel<HD, NCOL, true>
        <<<grid, 256, 0, stream>>>(Xh, Wdt, gates, Dg, nullptr);
  }
  // 4) GEMM2: out = Dg @ Wu          [f32 out, K=2048, N=1024]
  {
    dim3 grid(HD / 256, BS / 64);
    gemm_wmma_kernel<NCOL, HD, false>
        <<<grid, 256, 0, stream>>>(Dg, Wut, nullptr, nullptr, out);
  }
}
